// ConcatRelationModule_43980465111473
// MI455X (gfx1250) — matrix operand order repaired and checked by reading, not compiled, not run on hardware
//
#include <hip/hip_runtime.h>

typedef __attribute__((ext_vector_type(16))) _Float16 v16h;
typedef __attribute__((ext_vector_type(8)))  _Float16 v8h;
typedef __attribute__((ext_vector_type(8)))  float    v8f;

// Fragment-major f16 weight layout in d_ws (units: halves)
//   W1 = [WFOH | WFOM] as 256x256 : 8 kb x 16 nb tiles, 512 halves/tile
//   W2 = rhid2Layer     256x128   : 8 kb x  8 nb tiles
//   W3 = routLayer      128x 64   : 4 kb x  4 nb tiles
#define W1_OFF 0
#define W2_OFF 65536
#define W3_OFF 98304
#define W_TOTAL 106496

// ---------------------------------------------------------------------------
// Prep: convert f32 weights -> f16 fragments matching the WMMA B-operand
// layout (half j<8: K=8*(lane>>4)+j, j>=8: K=16+8*(lane>>4)+(j-8); N=lane&15).
// Element (tile t, lane l, half j) lives at t*512 + l*16 + j.
// ---------------------------------------------------------------------------
__global__ void prep_weights(const float* __restrict__ WFOH,
                             const float* __restrict__ WFOM,
                             const float* __restrict__ RH2,
                             const float* __restrict__ ROUT,
                             _Float16* __restrict__ w) {
  int idx = blockIdx.x * blockDim.x + threadIdx.x;
  if (idx >= W_TOTAL) return;
  int r   = idx & 511;
  int l   = r >> 4;
  int j   = r & 15;
  int kk  = (j < 8) ? ((l & 16) >> 1) + j : 16 + ((l & 16) >> 1) + (j - 8);   // K within the 32-row tile
  int n15 = l & 15;
  float v;
  if (idx < W2_OFF) {
    int t  = idx >> 9;              // 0..127
    int kb = t >> 4, nb = t & 15;
    int K  = kb * 32 + kk;
    int N  = nb * 16 + n15;
    v = (N < 128) ? WFOH[K * 128 + N] : WFOM[K * 128 + (N - 128)];
  } else if (idx < W3_OFF) {
    int t  = (idx - W2_OFF) >> 9;   // 0..63
    int kb = t >> 3, nb = t & 7;
    v = RH2[(kb * 32 + kk) * 128 + nb * 16 + n15];
  } else {
    int t  = (idx - W3_OFF) >> 9;   // 0..15
    int kb = t >> 2, nb = t & 3;
    v = ROUT[(kb * 32 + kk) * 64 + nb * 16 + n15];
  }
  w[idx] = (_Float16)v;
}

// ---------------------------------------------------------------------------
// Main kernel helpers
// ---------------------------------------------------------------------------
__device__ __forceinline__ v8f wmma16(v16h a, v16h b, v8f c) {
  return __builtin_amdgcn_wmma_f32_16x16x32_f16(
      /*neg_a=*/false, a, /*neg_b=*/false, b,
      /*c_mod=*/(short)0, c, /*reuse_a=*/false, /*reuse_b=*/false);
}

// Fast tanh: tanh(x) = 1 - 2/(exp2(2*log2(e)*x) + 1).
// exp2 -> single v_exp_f32; rcp -> single v_rcp_f32. Saturation falls out of
// exp2(+inf)=inf -> rcp=0 -> +1 and exp2(-inf)=0 -> 1-2 = -1.
__device__ __forceinline__ float fast_tanh(float x) {
  float e = exp2f(x * 2.8853900817779268f);   // 2/ln(2)
  return 1.0f - 2.0f * __builtin_amdgcn_rcpf(e + 1.0f);
}

// Build one A fragment (lane row, K runs [k0..k0+7] and [16+k0..16+k0+7]) from
// two 8-float f32 runs, converting to f16.
__device__ __forceinline__ v16h cvt16(const float* p0, const float* p1) {
  const float4 a0 = *(const float4*)(p0);
  const float4 a1 = *(const float4*)(p0 + 4);
  const float4 b0 = *(const float4*)(p1);
  const float4 b1 = *(const float4*)(p1 + 4);
  v16h r;
  r[0]  = (_Float16)a0.x; r[1]  = (_Float16)a0.y; r[2]  = (_Float16)a0.z; r[3]  = (_Float16)a0.w;
  r[4]  = (_Float16)a1.x; r[5]  = (_Float16)a1.y; r[6]  = (_Float16)a1.z; r[7]  = (_Float16)a1.w;
  r[8]  = (_Float16)b0.x; r[9]  = (_Float16)b0.y; r[10] = (_Float16)b0.z; r[11] = (_Float16)b0.w;
  r[12] = (_Float16)b1.x; r[13] = (_Float16)b1.y; r[14] = (_Float16)b1.z; r[15] = (_Float16)b1.w;
  return r;
}

#define HSTRIDE  264   // 256 + 8 pad halves (528B rows, 16B aligned)
#define H2STRIDE 136   // 128 + 8 pad halves (272B rows, 16B aligned)

__launch_bounds__(64)
__global__ void rel_label_kernel(const float* __restrict__ fwd,
                                 const float* __restrict__ bwd,
                                 const int* __restrict__ heads,
                                 const int* __restrict__ rels,
                                 const float* __restrict__ rcatBias,
                                 const float* __restrict__ rhid2Bias,
                                 const float* __restrict__ routBias,
                                 const _Float16* __restrict__ w,
                                 float* __restrict__ out,
                                 int E, int T) {
  __shared__ _Float16 hbuf[2][16 * HSTRIDE];
  __shared__ _Float16 h2buf[2][16 * H2STRIDE];

  const int lane = threadIdx.x & 31;
  const int wv   = threadIdx.x >> 5;
  const int tile = blockIdx.x * 2 + wv;
  if (tile >= T) return;

  const int l15 = lane & 15;
  const int hi  = lane >> 4;
  const int k0  = hi * 8;

  // ---- Gather: build all 8 A fragments of cat = [fwd[head] | bwd[mod]] ----
  int er = tile * 16 + l15;
  int ec = er < E ? er : E - 1;
  const float* rowF = fwd + (size_t)heads[ec] * 128;
  const float* rowB = bwd + (size_t)(ec + 1) * 128;
  v16h afrag[8];
#pragma unroll
  for (int kb = 0; kb < 8; ++kb) {
    const float* src = (kb < 4) ? (rowF + kb * 32) : (rowB + (kb - 4) * 32);
    afrag[kb] = cvt16(src + k0, src + 16 + k0);
  }

  const _Float16* w1 = w + W1_OFF;
  const _Float16* w2 = w + W2_OFF;
  const _Float16* w3 = w + W3_OFF;
  _Float16* hb  = hbuf[wv];
  _Float16* h2b = h2buf[wv];

  const v8f zero8 = {0.f, 0.f, 0.f, 0.f, 0.f, 0.f, 0.f, 0.f};

  // ---- GEMM1: (16x256) x (256x256) in two passes of 8 N-tiles ----
  for (int pass = 0; pass < 2; ++pass) {
    v8f acc[8];
#pragma unroll
    for (int n = 0; n < 8; ++n) acc[n] = zero8;
#pragma unroll
    for (int kb = 0; kb < 8; ++kb) {
#pragma unroll
      for (int n = 0; n < 8; ++n) {
        int nb = pass * 8 + n;
        v16h b = *(const v16h*)(w1 + (size_t)(kb * 16 + nb) * 512 + lane * 16);
        acc[n] = wmma16(afrag[kb], b, acc[n]);
      }
    }
#pragma unroll
    for (int n = 0; n < 8; ++n) {
      int col = (pass * 8 + n) * 16 + l15;
      float bias = rcatBias[col];
#pragma unroll
      for (int i = 0; i < 8; ++i) {
        float v = fast_tanh(acc[n][i] + bias);
        hb[(i + 8 * hi) * HSTRIDE + col] = (_Float16)v;
      }
    }
  }

  // ---- GEMM2: h (16x256) x rhid2 (256x128) ----
  v16h hf[8];
#pragma unroll
  for (int kb = 0; kb < 8; ++kb) {
    v8h lo = *(const v8h*)(hb + l15 * HSTRIDE + kb * 32 + k0);
    v8h hh = *(const v8h*)(hb + l15 * HSTRIDE + kb * 32 + 16 + k0);
    hf[kb] = __builtin_shufflevector(lo, hh, 0, 1, 2, 3, 4, 5, 6, 7,
                                     8, 9, 10, 11, 12, 13, 14, 15);
  }
  v8f acc2[8];
#pragma unroll
  for (int n = 0; n < 8; ++n) acc2[n] = zero8;
#pragma unroll
  for (int kb = 0; kb < 8; ++kb) {
#pragma unroll
    for (int n = 0; n < 8; ++n) {
      v16h b = *(const v16h*)(w2 + (size_t)(kb * 8 + n) * 512 + lane * 16);
      acc2[n] = wmma16(hf[kb], b, acc2[n]);
    }
  }
#pragma unroll
  for (int n = 0; n < 8; ++n) {
    int col = n * 16 + l15;
    float bias = rhid2Bias[col];
#pragma unroll
    for (int i = 0; i < 8; ++i) {
      float v = fast_tanh(acc2[n][i] + bias);
      h2b[(i + 8 * hi) * H2STRIDE + col] = (_Float16)v;
    }
  }

  // ---- GEMM3: h2 (16x128) x rout (128x64) ----
  v16h h2f[4];
#pragma unroll
  for (int kb = 0; kb < 4; ++kb) {
    v8h lo = *(const v8h*)(h2b + l15 * H2STRIDE + kb * 32 + k0);
    v8h hh = *(const v8h*)(h2b + l15 * H2STRIDE + kb * 32 + 16 + k0);
    h2f[kb] = __builtin_shufflevector(lo, hh, 0, 1, 2, 3, 4, 5, 6, 7,
                                      8, 9, 10, 11, 12, 13, 14, 15);
  }
  v8f acc3[4];
#pragma unroll
  for (int n = 0; n < 4; ++n) acc3[n] = zero8;
#pragma unroll
  for (int kb = 0; kb < 4; ++kb) {
#pragma unroll
    for (int n = 0; n < 4; ++n) {
      v16h b = *(const v16h*)(w3 + (size_t)(kb * 4 + n) * 512 + lane * 16);
      acc3[n] = wmma16(h2f[kb], b, acc3[n]);
    }
  }
  float rb[4];
#pragma unroll
  for (int n = 0; n < 4; ++n) rb[n] = routBias[n * 16 + l15];

  // ---- Hinge: per edge, gold score vs best wrong score across 64 labels ----
  const float NEGBIG = -3.0e38f;
#pragma unroll
  for (int i = 0; i < 8; ++i) {
    int e  = tile * 16 + i + 8 * hi;   // edge this (vgpr, half) row holds
    int eg = e < E ? e : E - 1;
    int g  = rels[eg];
    float gold = NEGBIG, wrong = NEGBIG;
#pragma unroll
    for (int n = 0; n < 4; ++n) {
      float s = acc3[n][i] + rb[n];
      int lbl = n * 16 + l15;
      if (lbl == g) gold = s;
      else          wrong = fmaxf(wrong, s);
    }
#pragma unroll
    for (int m = 1; m < 16; m <<= 1) {
      gold  = fmaxf(gold,  __shfl_xor(gold,  m, 32));
      wrong = fmaxf(wrong, __shfl_xor(wrong, m, 32));
    }
    if (l15 == 0 && e < E) {
      out[e] = (gold < wrong + 1.0f) ? (wrong - gold) : 0.0f;
    }
  }
}

// ---------------------------------------------------------------------------
extern "C" void kernel_launch(void* const* d_in, const int* in_sizes, int n_in,
                              void* d_out, int out_size, void* d_ws, size_t ws_size,
                              hipStream_t stream) {
  const float* fwd    = (const float*)d_in[0];
  const float* bwd    = (const float*)d_in[1];
  const int*   heads  = (const int*)d_in[2];
  const int*   rels   = (const int*)d_in[3];
  const float* WFOH   = (const float*)d_in[4];
  const float* WFOM   = (const float*)d_in[5];
  // d_in[6] rhidBias is unused by the reference
  const float* rcatB  = (const float*)d_in[7];
  const float* rhid2L = (const float*)d_in[8];
  const float* rhid2B = (const float*)d_in[9];
  const float* routL  = (const float*)d_in[10];
  const float* routB  = (const float*)d_in[11];

  _Float16* w = (_Float16*)d_ws;      // needs 212992 bytes
  int E = in_sizes[2];                // gold_heads has E = N-1 elements

  prep_weights<<<(W_TOTAL + 255) / 256, 256, 0, stream>>>(WFOH, WFOM, rhid2L,
                                                          routL, w);

  int T = (E + 15) / 16;              // 16-edge tiles
  int blocks = (T + 1) / 2;           // 2 waves (tiles) per 64-thread block
  rel_label_kernel<<<blocks, 64, 0, stream>>>(fwd, bwd, heads, rels, rcatB,
                                              rhid2B, routB, w, (float*)d_out,
                                              E, T);
}
